// LSTMSeq2Seq_67053029425634
// MI455X (gfx1250) — matrix operand order repaired and checked by reading, not compiled, not run on hardware
//
#include <hip/hip_runtime.h>

// ---------------------------------------------------------------------------
// Types for CDNA5 WMMA (wave32): v_wmma_f32_16x16x32_bf16
// ---------------------------------------------------------------------------
typedef __attribute__((ext_vector_type(16))) __bf16 v16bf;
typedef __attribute__((ext_vector_type(8)))  __bf16 v8bf;
typedef __attribute__((ext_vector_type(8)))  float  v8f;

#define B_SZ 256
#define T_SZ 128
#define F_SZ 64
#define H_SZ 512
#define G4H  (4 * H_SZ)   // 2048
#define O_SZ 16

// float -> bf16 round-to-nearest-even
__device__ __forceinline__ __bf16 f2bf(float f) {
    unsigned u = __builtin_bit_cast(unsigned, f);
    unsigned r = (u + 0x7FFFu + ((u >> 16) & 1u)) >> 16;
    unsigned short s = (unsigned short)r;
    return __builtin_bit_cast(__bf16, s);
}

__device__ __forceinline__ float sigmoidf_(float x) {
    return 1.0f / (1.0f + __expf(-x));
}
__device__ __forceinline__ float tanhf_(float x) {
    float e = __expf(2.0f * x);
    return (e - 1.0f) / (e + 1.0f);
}

// ---------------------------------------------------------------------------
// Fragment loaders
// A (16x32 bf16, M x K), row-major source with row stride lda (elements).
// Lane l holds row M = l%16; K-range [basek, basek+16)
// with basek = (l<16 ? 0 : 16).  Two 16-byte loads per lane.
// ---------------------------------------------------------------------------
__device__ __forceinline__ v16bf load_a_frag(const __bf16* __restrict__ A,
                                             int lda, int m0, int kk, int lane) {
    int m = m0 + (lane & 15);
    int basek = (lane < 16) ? 0 : 16;
    const __bf16* p = A + (size_t)m * lda + kk + basek;
    v8bf lo = *(const v8bf*)(p);
    v8bf hi = *(const v8bf*)(p + 8);
    v16bf a;
#pragma unroll
    for (int i = 0; i < 8; ++i) { a[i] = lo[i]; a[8 + i] = hi[i]; }
    return a;
}

// B (32x16 bf16, K x N) from pre-packed transposed weights Bp[N][K] (K contig).
// Lane l holds column N = l%16, K-range [kk + (l<16?0:16), +16): one v16bf load.
__device__ __forceinline__ v16bf load_b_frag(const __bf16* __restrict__ Bp,
                                             int Krow, int col0, int kk, int lane) {
    int n  = lane & 15;
    int kb = kk + ((lane < 16) ? 0 : 16);
    return *(const v16bf*)(Bp + (size_t)(col0 + n) * Krow + kb);
}

// ---------------------------------------------------------------------------
// Fused LSTM step: z = zinit + bias + A1@B1 + A2@B2  (gates i|f|g|o along N),
// then c' = sig(f)*c + sig(i)*tanh(g);  h = sig(o)*tanh(c').
// One wave (32 threads) per 16(batch) x 16(hidden) tile; 4 WMMA chains (gates).
// grid = (H/16, B/16), block = 32.  EXEC stays all-ones (uniform branches only).
// ---------------------------------------------------------------------------
__global__ __launch_bounds__(32) void lstm_step_kernel(
    const float* __restrict__ zinit, long long zStride,
    const float* __restrict__ bias,
    const __bf16* __restrict__ A1, int lda1, const __bf16* __restrict__ B1, int K1,
    const __bf16* __restrict__ A2, int lda2, const __bf16* __restrict__ B2, int K2,
    const float* __restrict__ c_in, float* __restrict__ c_out,
    __bf16* __restrict__ h_bf, float* __restrict__ h_f32)
{
    const int lane = threadIdx.x;
    const int n0   = blockIdx.x * 16;   // hidden-tile origin
    const int m0   = blockIdx.y * 16;   // batch-tile origin
    const int nl   = lane & 15;
    const int mofs = (lane >= 16) ? 8 : 0;

    v8f acc[4];
#pragma unroll
    for (int g = 0; g < 4; ++g) {
        float binit = bias ? bias[g * H_SZ + n0 + nl] : 0.0f;
#pragma unroll
        for (int r = 0; r < 8; ++r) {
            float z = binit;
            if (zinit) {
                int m = m0 + r + mofs;
                z += zinit[(size_t)m * zStride + g * H_SZ + n0 + nl];
            }
            acc[g][r] = z;
        }
    }

    // Recurrent path: h @ U  (K1 = 512)
    for (int kk = 0; kk < K1; kk += 32) {
        v16bf a = load_a_frag(A1, lda1, m0, kk, lane);
#pragma unroll
        for (int g = 0; g < 4; ++g) {
            v16bf b = load_b_frag(B1, K1, g * H_SZ + n0, kk, lane);
            acc[g] = __builtin_amdgcn_wmma_f32_16x16x32_bf16(
                false, a, false, b, (short)0, acc[g], false, false);
        }
    }
    // Optional input path: x @ W  (K2 = 64 or 512)
    if (A2) {
        for (int kk = 0; kk < K2; kk += 32) {
            v16bf a = load_a_frag(A2, lda2, m0, kk, lane);
#pragma unroll
            for (int g = 0; g < 4; ++g) {
                v16bf b = load_b_frag(B2, K2, g * H_SZ + n0, kk, lane);
                acc[g] = __builtin_amdgcn_wmma_f32_16x16x32_bf16(
                    false, a, false, b, (short)0, acc[g], false, false);
            }
        }
    }

    // Elementwise gates; C/D layout: VGPR r -> M = r (+8 for lanes 16..31)
#pragma unroll
    for (int r = 0; r < 8; ++r) {
        int m = m0 + r + mofs;
        size_t idx = (size_t)m * H_SZ + n0 + nl;
        float ig = sigmoidf_(acc[0][r]);
        float fg = sigmoidf_(acc[1][r]);
        float gg = tanhf_   (acc[2][r]);
        float og = sigmoidf_(acc[3][r]);
        float c  = fg * c_in[idx] + ig * gg;
        float h  = og * tanhf_(c);
        c_out[idx] = c;
        h_bf[idx]  = f2bf(h);
        if (h_f32) h_f32[idx] = h;
    }
}

// ---------------------------------------------------------------------------
// Helpers: pack fp32 [K][N] -> bf16 transposed [N][K]; fp32 -> bf16 copy; zero.
// ---------------------------------------------------------------------------
__global__ void pack_bf16_T(const float* __restrict__ in, __bf16* __restrict__ out,
                            int K, int N) {
    int idx = blockIdx.x * blockDim.x + threadIdx.x;
    if (idx >= K * N) return;
    int k = idx / N, n = idx - k * N;
    out[(size_t)n * K + k] = f2bf(in[idx]);
}
__global__ void cvt_bf16(const float* __restrict__ in, __bf16* __restrict__ out, int n) {
    int idx = blockIdx.x * blockDim.x + threadIdx.x;
    if (idx < n) out[idx] = f2bf(in[idx]);
}
__global__ void zero_f32(float* p, int n) {
    int idx = blockIdx.x * blockDim.x + threadIdx.x;
    if (idx < n) p[idx] = 0.0f;
}
__global__ void zero_bf16(__bf16* p, int n) {
    int idx = blockIdx.x * blockDim.x + threadIdx.x;
    if (idx < n) p[idx] = f2bf(0.0f);
}

// Decoder input GEMM (tiny, K=16): z[b][g] = b0[g] + dec_in[b,:16] . W0[:16,g]
__global__ void dec_xw_kernel(const float* __restrict__ inp, const float* __restrict__ W,
                              const float* __restrict__ bias, float* __restrict__ z) {
    int idx = blockIdx.x * blockDim.x + threadIdx.x;
    if (idx >= B_SZ * G4H) return;
    int b = idx >> 11, g = idx & (G4H - 1);
    const float* xr = inp + (size_t)b * T_SZ * F_SZ + (size_t)(T_SZ - 1) * F_SZ;
    float s = bias[g];
#pragma unroll
    for (int k = 0; k < O_SZ; ++k) s += xr[k] * W[k * G4H + g];
    z[idx] = s;
}

// FC + tile to future_seq_len: out[b,t,o] = h[b,:] . fcW[:,o] + fcb[o]
__global__ void fc_tile_kernel(const float* __restrict__ h, const float* __restrict__ W,
                               const float* __restrict__ bias, float* __restrict__ out,
                               int fut) {
    int idx = blockIdx.x * blockDim.x + threadIdx.x;
    if (idx >= B_SZ * O_SZ) return;
    int b = idx >> 4, o = idx & 15;
    float s = bias[o];
    const float* hr = h + (size_t)b * H_SZ;
    for (int k = 0; k < H_SZ; ++k) s += hr[k] * W[k * O_SZ + o];
    for (int t = 0; t < fut; ++t)
        out[(size_t)b * fut * O_SZ + (size_t)t * O_SZ + o] = s;
}

// ---------------------------------------------------------------------------
extern "C" void kernel_launch(void* const* d_in, const int* in_sizes, int n_in,
                              void* d_out, int out_size, void* d_ws, size_t ws_size,
                              hipStream_t stream) {
    const float* inp   = (const float*)d_in[0];
    const float* encW0 = (const float*)d_in[2];
    const float* encU0 = (const float*)d_in[3];
    const float* encb0 = (const float*)d_in[4];
    const float* encW1 = (const float*)d_in[5];
    const float* encU1 = (const float*)d_in[6];
    const float* encb1 = (const float*)d_in[7];
    const float* decW0 = (const float*)d_in[8];
    const float* decU0 = (const float*)d_in[9];
    const float* decb0 = (const float*)d_in[10];
    const float* decW1 = (const float*)d_in[11];
    const float* decU1 = (const float*)d_in[12];
    const float* decb1 = (const float*)d_in[13];
    const float* fcW   = (const float*)d_in[14];
    const float* fcb   = (const float*)d_in[15];
    float* out = (float*)d_out;
    const int fut = out_size / (B_SZ * O_SZ);   // == future_seq_len (48)

    // ---- workspace carve-out (256B aligned) ----
    char* ws = (char*)d_ws;
    size_t off = 0;
    auto alloc = [&](size_t bytes) -> void* {
        off = (off + 255) & ~(size_t)255;
        void* p = ws + off;
        off += bytes;
        return p;
    };
    __bf16* xb   = (__bf16*)alloc((size_t)B_SZ * T_SZ * F_SZ * 2);       // bf16 input
    __bf16* W0p  = (__bf16*)alloc((size_t)G4H * F_SZ * 2);               // packed [2048][64]
    __bf16* U0p  = (__bf16*)alloc((size_t)G4H * H_SZ * 2);               // packed [2048][512]
    __bf16* W1p  = (__bf16*)alloc((size_t)G4H * H_SZ * 2);
    __bf16* U1p  = (__bf16*)alloc((size_t)G4H * H_SZ * 2);
    __bf16* dU0p = (__bf16*)alloc((size_t)G4H * H_SZ * 2);
    __bf16* dW1p = (__bf16*)alloc((size_t)G4H * H_SZ * 2);
    __bf16* dU1p = (__bf16*)alloc((size_t)G4H * H_SZ * 2);
    __bf16* h0a  = (__bf16*)alloc((size_t)B_SZ * H_SZ * 2);
    __bf16* h0b  = (__bf16*)alloc((size_t)B_SZ * H_SZ * 2);
    __bf16* h1a  = (__bf16*)alloc((size_t)B_SZ * H_SZ * 2);
    __bf16* h1b  = (__bf16*)alloc((size_t)B_SZ * H_SZ * 2);
    __bf16* hd0  = (__bf16*)alloc((size_t)B_SZ * H_SZ * 2);
    __bf16* hd1b = (__bf16*)alloc((size_t)B_SZ * H_SZ * 2);
    float*  c0   = (float*) alloc((size_t)B_SZ * H_SZ * 4);
    float*  c1   = (float*) alloc((size_t)B_SZ * H_SZ * 4);
    float*  cd0  = (float*) alloc((size_t)B_SZ * H_SZ * 4);
    float*  cd1  = (float*) alloc((size_t)B_SZ * H_SZ * 4);
    float*  hd1f = (float*) alloc((size_t)B_SZ * H_SZ * 4);
    float*  zdec = (float*) alloc((size_t)B_SZ * G4H * 4);
    (void)ws_size; (void)n_in; (void)in_sizes;

    const int TPB = 256;
    auto blk = [](int n, int t) { return (n + t - 1) / t; };

    // ---- one-time per-call prep ----
    int nx = B_SZ * T_SZ * F_SZ;
    cvt_bf16<<<blk(nx, TPB), TPB, 0, stream>>>(inp, xb, nx);
    pack_bf16_T<<<blk(F_SZ * G4H, TPB), TPB, 0, stream>>>(encW0, W0p, F_SZ, G4H);
    pack_bf16_T<<<blk(H_SZ * G4H, TPB), TPB, 0, stream>>>(encU0, U0p, H_SZ, G4H);
    pack_bf16_T<<<blk(H_SZ * G4H, TPB), TPB, 0, stream>>>(encW1, W1p, H_SZ, G4H);
    pack_bf16_T<<<blk(H_SZ * G4H, TPB), TPB, 0, stream>>>(encU1, U1p, H_SZ, G4H);
    pack_bf16_T<<<blk(H_SZ * G4H, TPB), TPB, 0, stream>>>(decU0, dU0p, H_SZ, G4H);
    pack_bf16_T<<<blk(H_SZ * G4H, TPB), TPB, 0, stream>>>(decW1, dW1p, H_SZ, G4H);
    pack_bf16_T<<<blk(H_SZ * G4H, TPB), TPB, 0, stream>>>(decU1, dU1p, H_SZ, G4H);
    int ns = B_SZ * H_SZ;
    zero_f32 <<<blk(ns, TPB), TPB, 0, stream>>>(c0, ns);
    zero_f32 <<<blk(ns, TPB), TPB, 0, stream>>>(c1, ns);
    zero_bf16<<<blk(ns, TPB), TPB, 0, stream>>>(h0a, ns);
    zero_bf16<<<blk(ns, TPB), TPB, 0, stream>>>(h1a, ns);

    // ---- encoder: 2 stacked LSTM layers, fused per timestep ----
    dim3 sgrid(H_SZ / 16, B_SZ / 16);   // 32 x 16 waves
    __bf16 *h0c = h0a, *h0n = h0b, *h1c = h1a, *h1n = h1b;
    for (int t = 0; t < T_SZ; ++t) {
        // layer 0: z = b0 + h0@U0 + x_t@W0
        lstm_step_kernel<<<sgrid, 32, 0, stream>>>(
            nullptr, 0, encb0,
            h0c, H_SZ, U0p, H_SZ,
            xb + (size_t)t * F_SZ, T_SZ * F_SZ, W0p, F_SZ,
            c0, c0, h0n, nullptr);
        { __bf16* tmp = h0c; h0c = h0n; h0n = tmp; }
        // layer 1: z = b1 + h1@U1 + h0_t@W1
        lstm_step_kernel<<<sgrid, 32, 0, stream>>>(
            nullptr, 0, encb1,
            h1c, H_SZ, U1p, H_SZ,
            h0c, H_SZ, W1p, H_SZ,
            c1, c1, h1n, nullptr);
        { __bf16* tmp = h1c; h1c = h1n; h1n = tmp; }
    }

    // ---- decoder (single step each layer), init state = (h1, c1) ----
    dec_xw_kernel<<<blk(B_SZ * G4H, TPB), TPB, 0, stream>>>(inp, decW0, decb0, zdec);
    // layer d0: z = zdec (incl. bias) + h1@dec_U0 ; c_in = c1 -> cd0
    lstm_step_kernel<<<sgrid, 32, 0, stream>>>(
        zdec, G4H, nullptr,
        h1c, H_SZ, dU0p, H_SZ,
        nullptr, 0, nullptr, 0,
        c1, cd0, hd0, nullptr);
    // layer d1: z = b1 + hd0@dec_U1 + hd0@dec_W1 ; c_in = cd0 -> cd1
    lstm_step_kernel<<<sgrid, 32, 0, stream>>>(
        nullptr, 0, decb1,
        hd0, H_SZ, dU1p, H_SZ,
        hd0, H_SZ, dW1p, H_SZ,
        cd0, cd1, hd1b, hd1f);

    // ---- FC + tile over future_seq_len ----
    fc_tile_kernel<<<blk(B_SZ * O_SZ, TPB), TPB, 0, stream>>>(hd1f, fcW, fcb, out, fut);
}
